// LSTMEncoder1_46660524704393
// MI455X (gfx1250) — compile-verified, matrix operand order checked by reading, not run on hardware
//
#include <hip/hip_runtime.h>
#include <hip/hip_bf16.h>

// ---------------- problem constants ----------------
#define BB     32
#define TT     512
#define IN_SZ  1024
#define H_SZ   1024
#define G4     4096              // 4*H
#define MTOT   (BB * TT)         // 16384 rows for the input-projection GEMM

typedef __attribute__((ext_vector_type(16))) __bf16 v16bf;
typedef __attribute__((ext_vector_type(8)))  float  v8f;

union Frag {                     // one WMMA bf16 A/B operand (16 bf16 per lane)
    v16bf v;
    uint4 q[2];
};

// ---- CDNA5 async global->LDS staging (ASYNCcnt path), guarded ----
#if defined(__has_builtin)
#  if __has_builtin(__builtin_amdgcn_global_load_async_to_lds_b128) && \
      __has_builtin(__builtin_amdgcn_s_wait_asynccnt)
#    define USE_ASYNC_LDS 1
#  endif
#endif
#ifndef USE_ASYNC_LDS
#  define USE_ASYNC_LDS 0
#endif

// builtin wants pointers to a 4 x i32 generic vector (per hipcc diagnostic)
typedef int b128_t __attribute__((vector_size(16)));
typedef __attribute__((address_space(1))) b128_t* g128p;
typedef __attribute__((address_space(3))) b128_t* l128p;

__device__ __forceinline__ void stage_b128(const unsigned short* g, unsigned short* l) {
#if USE_ASYNC_LDS
    // per-lane 16B: global -> LDS directly, tracked by ASYNCcnt (no VGPR round-trip)
    __builtin_amdgcn_global_load_async_to_lds_b128(
        (g128p)(unsigned long long)g,      // as1 = full VA, int->ptr is exact
        (l128p)(void*)l,                   // addrspacecast flat -> LDS
        0, 0);
#else
    *(uint4*)l = *(const uint4*)g;
#endif
}
__device__ __forceinline__ void async_wait(int pending_next) {
#if USE_ASYNC_LDS
    if (pending_next) __builtin_amdgcn_s_wait_asynccnt(4);  // own 4 next-tile loads may stay in flight
    else              __builtin_amdgcn_s_wait_asynccnt(0);
#else
    (void)pending_next;
#endif
}

__device__ __forceinline__ unsigned short f32_to_bf16_rne(float f) {
    unsigned int u = __float_as_uint(f);
    u += 0x7fffu + ((u >> 16) & 1u);     // round-to-nearest-even
    return (unsigned short)(u >> 16);
}
__device__ __forceinline__ float sigm(float x) { return 1.0f / (1.0f + __expf(-x)); }

// ---------------- conversion kernels ----------------
// x fp32 [B,T,IN] -> bf16 row-major [MTOT, IN]
__global__ __launch_bounds__(256) void convert_x_kernel(const float* __restrict__ x,
                                                        unsigned short* __restrict__ xbf,
                                                        int total) {
    int i = blockIdx.x * 256 + threadIdx.x;
    if (i < total) xbf[i] = f32_to_bf16_rne(x[i]);
}

// W fp32 [2048, 4096] -> transposed bf16 WxT [4096,1024], WhT [4096,1024]
__global__ __launch_bounds__(256) void convert_w_kernel(const float* __restrict__ W,
                                                        unsigned short* __restrict__ WxT,
                                                        unsigned short* __restrict__ WhT) {
    int idx = blockIdx.x * 256 + threadIdx.x;       // 4096*1024 threads
    if (idx >= G4 * IN_SZ) return;
    int n = idx >> 10;          // output column of original W (0..4095)
    int k = idx & 1023;         // reduction index (0..1023)
    WxT[idx] = f32_to_bf16_rne(W[k * G4 + n]);
    WhT[idx] = f32_to_bf16_rne(W[(IN_SZ + k) * G4 + n]);
}

// ---------------- big input-projection GEMM ----------------
// C[MTOT, 4096] = A[MTOT,1024]bf16 * Bt[4096,1024]^T bf16   (fp32 accumulate)
// Block tile 128x128, 8 waves (4x2), wave tile 32x64, K-tile 32.
// Double-buffered LDS staged via async global->LDS (overlaps fetch k+1 with WMMA k).
__global__ __launch_bounds__(256) void gemm_xproj_kernel(const unsigned short* __restrict__ A,
                                                         const unsigned short* __restrict__ Bt,
                                                         float* __restrict__ C) {
    __shared__ unsigned short As[2][128 * 32];
    __shared__ unsigned short Bs[2][128 * 32];

    const int tid  = threadIdx.x;
    const int lane = tid & 31;
    const int wid  = tid >> 5;
    const int wm   = wid >> 1;          // 0..3 (M)
    const int wn   = wid & 1;           // 0..1 (N)
    const int bm   = blockIdx.y;        // 0..127
    const int bn   = blockIdx.x;        // 0..31
    const int l15  = lane & 15;
    const int hi   = (lane >> 4) & 1;

    const int rA    = tid >> 1;         // 0..127: row of the staged tiles
    const int halfk = (tid & 1) * 16;   // element offset within the 32-wide K tile

    const unsigned short* gA = A  + (bm * 128 + rA) * IN_SZ + halfk;
    const unsigned short* gB = Bt + (bn * 128 + rA) * IN_SZ + halfk;

    const v8f z8 = {0.f, 0.f, 0.f, 0.f, 0.f, 0.f, 0.f, 0.f};
    v8f acc[2][4];
    for (int i = 0; i < 2; ++i)
        for (int j = 0; j < 4; ++j) acc[i][j] = z8;

    // prologue: stage K-tile 0 into buffer 0
    {
        stage_b128(gA,     &As[0][rA * 32 + halfk]);
        stage_b128(gA + 8, &As[0][rA * 32 + halfk + 8]);
        stage_b128(gB,     &Bs[0][rA * 32 + halfk]);
        stage_b128(gB + 8, &Bs[0][rA * 32 + halfk + 8]);
    }

    for (int it = 0; it < IN_SZ / 32; ++it) {
        const int  cur  = it & 1;
        const bool more = (it + 1) < (IN_SZ / 32);
        if (more) {                      // issue next tile into the other buffer
            const int ko = (it + 1) * 32;
            stage_b128(gA + ko,     &As[cur ^ 1][rA * 32 + halfk]);
            stage_b128(gA + ko + 8, &As[cur ^ 1][rA * 32 + halfk + 8]);
            stage_b128(gB + ko,     &Bs[cur ^ 1][rA * 32 + halfk]);
            stage_b128(gB + ko + 8, &Bs[cur ^ 1][rA * 32 + halfk + 8]);
            __builtin_prefetch(gA + ko + 32, 0, 1);
            __builtin_prefetch(gB + ko + 32, 0, 1);
        }
        async_wait(more ? 1 : 0);        // current tile's own loads complete
        __syncthreads();                 // everyone's current tile visible in LDS

        Frag a[2], b[4];
        #pragma unroll
        for (int mi = 0; mi < 2; ++mi) {
            const unsigned short* p = &As[cur][(wm * 32 + mi * 16 + l15) * 32 + hi * 8];
            a[mi].q[0] = *(const uint4*)(p);
            a[mi].q[1] = *(const uint4*)(p + 16);
        }
        #pragma unroll
        for (int ni = 0; ni < 4; ++ni) {
            const unsigned short* p = &Bs[cur][(wn * 64 + ni * 16 + l15) * 32 + hi * 8];
            b[ni].q[0] = *(const uint4*)(p);
            b[ni].q[1] = *(const uint4*)(p + 16);
        }
        #pragma unroll
        for (int mi = 0; mi < 2; ++mi)
            #pragma unroll
            for (int ni = 0; ni < 4; ++ni)
                acc[mi][ni] = __builtin_amdgcn_wmma_f32_16x16x32_bf16(
                    false, a[mi].v, false, b[ni].v, (short)0, acc[mi][ni], false, false);

        __syncthreads();                 // WAR: buffer being refilled next iter
    }

    #pragma unroll
    for (int mi = 0; mi < 2; ++mi)
        #pragma unroll
        for (int ni = 0; ni < 4; ++ni) {
            const int col  = bn * 128 + wn * 64 + ni * 16 + l15;
            const int row0 = bm * 128 + wm * 32 + mi * 16 + hi * 8;
            #pragma unroll
            for (int r = 0; r < 8; ++r)
                C[(row0 + r) * G4 + col] = acc[mi][ni][r];
        }
}

// ---------------- one recurrent LSTM timestep ----------------
// Z = Xproj[t] + h_prev @ Wh + b ; gate order (i, j, f, o), forget bias on f.
// Grid: 8 blocks x 256 threads; block owns 128 hidden units; wave owns 16 units
// and computes all 4 gates for both 16-row batch tiles => 8 WMMA per K-iter.
__global__ __launch_bounds__(256) void lstm_step_kernel(
        const float* __restrict__ Xp,             // [MTOT, 4096]
        const unsigned short* __restrict__ WhT,   // [4096, 1024] bf16 transposed
        const float* __restrict__ bias,           // [4096]
        const int*  __restrict__ lengths,         // [32]
        const unsigned short* __restrict__ hbf_in, // [32,1024] bf16 (prev h)
        unsigned short* __restrict__ hbf_out,      // [32,1024] bf16 (next h)
        float* __restrict__ h_state,               // [32,1024] fp32
        float* __restrict__ c_state,               // [32,1024] fp32
        float* __restrict__ outF,                  // [MTOT,1024] or null
        unsigned short* __restrict__ outBf,        // [MTOT,1024] bf16 or null
        int t) {
    const int lane = threadIdx.x & 31;
    const int wid  = threadIdx.x >> 5;
    const int l15  = lane & 15;
    const int hi   = (lane >> 4) & 1;
    const int n0   = blockIdx.x * 128 + wid * 16;   // this wave's hidden-unit base

    const v8f z8 = {0.f, 0.f, 0.f, 0.f, 0.f, 0.f, 0.f, 0.f};
    v8f acc[4][2];                                  // [gate][batch-tile]
    for (int g = 0; g < 4; ++g)
        for (int mi = 0; mi < 2; ++mi) acc[g][mi] = z8;

    for (int kt = 0; kt < H_SZ; kt += 32) {
        Frag a[2], b[4];
        #pragma unroll
        for (int mi = 0; mi < 2; ++mi) {
            const unsigned short* p = hbf_in + (mi * 16 + l15) * H_SZ + kt + hi * 8;
            a[mi].q[0] = *(const uint4*)(p);
            a[mi].q[1] = *(const uint4*)(p + 16);
        }
        #pragma unroll
        for (int g = 0; g < 4; ++g) {
            const unsigned short* p = WhT + (g * H_SZ + n0 + l15) * H_SZ + kt + hi * 8;
            b[g].q[0] = *(const uint4*)(p);
            b[g].q[1] = *(const uint4*)(p + 16);
        }
        if (kt + 32 < H_SZ)
            __builtin_prefetch(WhT + (n0 + l15) * H_SZ + kt + 32, 0, 1);
        #pragma unroll
        for (int g = 0; g < 4; ++g)
            #pragma unroll
            for (int mi = 0; mi < 2; ++mi)
                acc[g][mi] = __builtin_amdgcn_wmma_f32_16x16x32_bf16(
                    false, a[mi].v, false, b[g].v, (short)0, acc[g][mi], false, false);
    }

    // Epilogue: pointwise LSTM cell per (batch, unit) owned by this lane.
    const int n  = n0 + l15;
    const float bi  = bias[n];
    const float bj  = bias[H_SZ + n];
    const float bff = bias[2 * H_SZ + n] + 1.0f;    // FORGET_BIAS
    const float bo  = bias[3 * H_SZ + n];

    #pragma unroll
    for (int mi = 0; mi < 2; ++mi) {
        #pragma unroll
        for (int r = 0; r < 8; ++r) {
            const int bidx = mi * 16 + hi * 8 + r;           // batch index
            const float* xp = Xp + (size_t)(bidx * TT + t) * G4;
            const float zi = acc[0][mi][r] + xp[n]              + bi;
            const float zj = acc[1][mi][r] + xp[H_SZ + n]       + bj;
            const float zf = acc[2][mi][r] + xp[2 * H_SZ + n]   + bff;
            const float zo = acc[3][mi][r] + xp[3 * H_SZ + n]   + bo;
            const float cprev = c_state[bidx * H_SZ + n];
            const float cn = sigm(zf) * cprev + sigm(zi) * tanhf(zj);
            const float hn = sigm(zo) * tanhf(cn);
            const bool  m  = (t < lengths[bidx]);
            const float hk = m ? hn : h_state[bidx * H_SZ + n];
            const float ck = m ? cn : cprev;
            h_state[bidx * H_SZ + n] = hk;
            c_state[bidx * H_SZ + n] = ck;
            hbf_out[bidx * H_SZ + n] = f32_to_bf16_rne(hk);
            const float ho = m ? hn : 0.0f;
            if (outF)  outF [(size_t)(bidx * TT + t) * H_SZ + n] = ho;
            if (outBf) outBf[(size_t)(bidx * TT + t) * H_SZ + n] = f32_to_bf16_rne(ho);
        }
    }
}

// Copy final (h1, c1) to the tail of d_out.
__global__ __launch_bounds__(256) void write_states_kernel(const float* __restrict__ h_state,
                                                           const float* __restrict__ c_state,
                                                           float* __restrict__ out_tail) {
    int i = blockIdx.x * 256 + threadIdx.x;
    if (i < BB * H_SZ) {
        out_tail[i]              = h_state[i];
        out_tail[BB * H_SZ + i]  = c_state[i];
    }
}

// ---------------- host orchestration ----------------
extern "C" void kernel_launch(void* const* d_in, const int* in_sizes, int n_in,
                              void* d_out, int out_size, void* d_ws, size_t ws_size,
                              hipStream_t stream) {
    (void)in_sizes; (void)n_in; (void)out_size; (void)ws_size;
    const float* x       = (const float*)d_in[0];
    const int*   lengths = (const int*)  d_in[1];
    const float* W0      = (const float*)d_in[2];
    const float* b0      = (const float*)d_in[3];
    const float* W1      = (const float*)d_in[4];
    const float* b1      = (const float*)d_in[5];
    float* out = (float*)d_out;

    // workspace carve (deterministic)
    char* ws = (char*)d_ws;
    size_t off = 0;
    auto carve = [&](size_t bytes) -> void* {
        void* p = ws + off;
        off += (bytes + 255) & ~(size_t)255;
        return p;
    };
    float*          Xp   = (float*)         carve((size_t)MTOT * G4 * sizeof(float));   // 256 MB
    unsigned short* xbf0 = (unsigned short*)carve((size_t)MTOT * IN_SZ * 2);            // 32 MB
    unsigned short* xbf1 = (unsigned short*)carve((size_t)MTOT * H_SZ  * 2);            // 32 MB
    unsigned short* WxT0 = (unsigned short*)carve((size_t)G4 * IN_SZ * 2);              // 8 MB
    unsigned short* WhT0 = (unsigned short*)carve((size_t)G4 * H_SZ  * 2);
    unsigned short* WxT1 = (unsigned short*)carve((size_t)G4 * H_SZ  * 2);
    unsigned short* WhT1 = (unsigned short*)carve((size_t)G4 * H_SZ  * 2);
    unsigned short* hbfA = (unsigned short*)carve((size_t)BB * H_SZ * 2);
    unsigned short* hbfB = (unsigned short*)carve((size_t)BB * H_SZ * 2);
    float*          hSt  = (float*)         carve((size_t)BB * H_SZ * sizeof(float));
    float*          cSt  = (float*)         carve((size_t)BB * H_SZ * sizeof(float));

    // 1) precision conversion / weight transpose (one-time, parallel)
    convert_x_kernel<<<(MTOT * IN_SZ + 255) / 256, 256, 0, stream>>>(x, xbf0, MTOT * IN_SZ);
    convert_w_kernel<<<(G4 * IN_SZ + 255) / 256, 256, 0, stream>>>(W0, WxT0, WhT0);
    convert_w_kernel<<<(G4 * H_SZ + 255) / 256, 256, 0, stream>>>(W1, WxT1, WhT1);

    dim3 ggrid(G4 / 128, MTOT / 128);   // 32 x 128 blocks

    // ---------------- layer 0 ----------------
    (void)hipMemsetAsync(hbfA, 0, (size_t)BB * H_SZ * 2, stream);
    (void)hipMemsetAsync(hbfB, 0, (size_t)BB * H_SZ * 2, stream);
    (void)hipMemsetAsync(hSt, 0, (size_t)BB * H_SZ * sizeof(float), stream);
    (void)hipMemsetAsync(cSt, 0, (size_t)BB * H_SZ * sizeof(float), stream);
    gemm_xproj_kernel<<<ggrid, 256, 0, stream>>>(xbf0, WxT0, Xp);
    for (int t = 0; t < TT; ++t) {
        unsigned short* hprev = (t & 1) ? hbfB : hbfA;
        unsigned short* hnext = (t & 1) ? hbfA : hbfB;
        lstm_step_kernel<<<H_SZ / 128, 256, 0, stream>>>(
            Xp, WhT0, b0, lengths, hprev, hnext, hSt, cSt,
            /*outF=*/nullptr, /*outBf=*/xbf1, t);
    }

    // ---------------- layer 1 ----------------
    (void)hipMemsetAsync(hbfA, 0, (size_t)BB * H_SZ * 2, stream);
    (void)hipMemsetAsync(hbfB, 0, (size_t)BB * H_SZ * 2, stream);
    (void)hipMemsetAsync(hSt, 0, (size_t)BB * H_SZ * sizeof(float), stream);
    (void)hipMemsetAsync(cSt, 0, (size_t)BB * H_SZ * sizeof(float), stream);
    gemm_xproj_kernel<<<ggrid, 256, 0, stream>>>(xbf1, WxT1, Xp);
    for (int t = 0; t < TT; ++t) {
        unsigned short* hprev = (t & 1) ? hbfB : hbfA;
        unsigned short* hnext = (t & 1) ? hbfA : hbfB;
        lstm_step_kernel<<<H_SZ / 128, 256, 0, stream>>>(
            Xp, WhT1, b1, lengths, hprev, hnext, hSt, cSt,
            /*outF=*/out, /*outBf=*/nullptr, t);
    }

    // 3) final states (h1, c1) appended after out1
    write_states_kernel<<<(BB * H_SZ + 255) / 256, 256, 0, stream>>>(
        hSt, cSt, out + (size_t)MTOT * H_SZ);
}
